// WKVAttentionChunk_68805376082236
// MI455X (gfx1250) — matrix operand order repaired and checked by reading, not compiled, not run on hardware
//
#include <hip/hip_runtime.h>
#include <hip/hip_bf16.h>

typedef __bf16 bf16;
typedef __attribute__((ext_vector_type(16))) bf16  v16bf;
typedef __attribute__((ext_vector_type(8)))  float v8f;
typedef __attribute__((ext_vector_type(4)))  unsigned int v4u;

#define B_  8
#define T_  4096
#define C_  512
#define H_  8
#define D_  64
#define BT_ (B_*T_)

__device__ __forceinline__ float sigm(float x) { return 1.0f / (1.0f + __expf(-x)); }

// ---------------------------------------------------------------- f32 -> bf16
__global__ void cvt_bf16_kernel(const float* __restrict__ src, bf16* __restrict__ dst, int n) {
    int i = blockIdx.x * 256 + threadIdx.x;
    if (i < n) dst[i] = (bf16)src[i];
}

// ------------------------------------------- bqe time-shift mix + bf16 cast (x5)
__global__ void bqe_cast_kernel(const float* __restrict__ x,
                                const float* __restrict__ mur, const float* __restrict__ muk,
                                const float* __restrict__ muv, const float* __restrict__ muw,
                                bf16* __restrict__ xr, bf16* __restrict__ xk,
                                bf16* __restrict__ xv, bf16* __restrict__ xw,
                                bf16* __restrict__ xg) {
    int gid = blockIdx.x * 256 + threadIdx.x;          // over BT_*C_
    int c   = gid & (C_ - 1);
    int bt  = gid >> 9;
    int t   = bt & (T_ - 1);
    int b   = bt >> 12;
    int q   = c >> 7;                                   // channel quarter
    int s   = (q == 0) ? 0 : (q == 1) ? 1 : (q == 2) ? -1 : 2;  // roll shift
    int ts  = (t - s) & (T_ - 1);                       // wrap like jnp.roll
    float xc = x[gid];
    float xs = x[(((b << 12) + ts) << 9) + c];
    xr[gid] = (bf16)(xc + (1.0f - sigm(mur[c])) * xs);
    xk[gid] = (bf16)(xc + (1.0f - sigm(muk[c])) * xs);
    xv[gid] = (bf16)(xc + (1.0f - sigm(muv[c])) * xs);
    xw[gid] = (bf16)(xc + (1.0f - sigm(muw[c])) * xs);
    xg[gid] = (bf16)xc;
}

// ------------------------------------------------- WMMA bf16 GEMM: out = A @ W^T
// A: (BT_ x C_) bf16 row-major, W: (C_ x C_) bf16 row-major (out[m][n] = sum_k A[m][k]*W[n][k])
union FragBF { v4u u[2]; v16bf b; };

__global__ void __launch_bounds__(256, 2)
gemm_bf16_kernel(const bf16* __restrict__ A, const bf16* __restrict__ W,
                 float* __restrict__ out) {
    const int lane = threadIdx.x & 31;
    const int wave = threadIdx.x >> 5;
    const int lh   = lane & 15;          // row/col within tile
    const int lx   = lane >> 4;          // half-wave select
    const int m0   = blockIdx.x * 128 + (wave >> 1) * 32;   // wave: 32 rows
    const int n0   = blockIdx.y * 128 + (wave & 1) * 64;    // wave: 64 cols

    v8f acc[2][4];
#pragma unroll
    for (int mi = 0; mi < 2; ++mi)
#pragma unroll
        for (int ni = 0; ni < 4; ++ni) acc[mi][ni] = (v8f)(0.0f);

    for (int kk = 0; kk < C_; kk += 32) {
        FragBF a[2], bb[4];
#pragma unroll
        for (int mi = 0; mi < 2; ++mi) {
            // A 16x32 bf16 frag: lanes 0-15 K{0..7,16..23}, lanes 16-31 K{8..15,24..31}
            const bf16* ap = A + (m0 + mi * 16 + lh) * C_ + kk + lx * 8;
            a[mi].u[0] = *(const v4u*)(ap);
            a[mi].u[1] = *(const v4u*)(ap + 16);
        }
#pragma unroll
        for (int ni = 0; ni < 4; ++ni) {
            // B 32x16 frag: lane = column n (row of W); lanes 0-15 K{0..7,16..23}, 16-31 K{8..15,24..31}
            const bf16* bp = W + (n0 + ni * 16 + lh) * C_ + kk + lx * 8;
            bb[ni].u[0] = *(const v4u*)(bp);
            bb[ni].u[1] = *(const v4u*)(bp + 16);
        }
#pragma unroll
        for (int mi = 0; mi < 2; ++mi)
#pragma unroll
            for (int ni = 0; ni < 4; ++ni)
                acc[mi][ni] = __builtin_amdgcn_wmma_f32_16x16x32_bf16(
                    false, a[mi].b, false, bb[ni].b, (short)0, acc[mi][ni], false, false);
    }
    // C/D layout: lane 0-15 -> N=lane, M=vgpr; lane 16-31 -> N=lane-16, M=8+vgpr
#pragma unroll
    for (int mi = 0; mi < 2; ++mi)
#pragma unroll
        for (int ni = 0; ni < 4; ++ni) {
            int col = n0 + ni * 16 + lh;
            int row = m0 + mi * 16 + lx * 8;
#pragma unroll
            for (int v = 0; v < 8; ++v)
                out[(row + v) * C_ + col] = acc[mi][ni][v];
        }
}

// -------------------------------------------------------- bidirectional WKV scan
__global__ void __launch_bounds__(64)
wkv_scan_kernel(const float* __restrict__ r, const float* __restrict__ k,
                const float* __restrict__ v, const float* __restrict__ cpr,
                const float* __restrict__ dA, const float* __restrict__ dB,
                const float* __restrict__ dbase, const float* __restrict__ u,
                float* __restrict__ y) {
    const int b  = blockIdx.x >> 3;
    const int h  = blockIdx.x & 7;
    const int d  = threadIdx.x;
    const int ch = h * D_ + d;
    const float uv = u[ch], aA = dA[ch], aB = dB[ch], ab = dbase[ch];

    __shared__ float kb[D_], vb[D_];
    float st[D_];

    for (int dir = 0; dir < 2; ++dir) {
#pragma unroll
        for (int e = 0; e < D_; ++e) st[e] = 0.0f;
        for (int i = 0; i < T_; ++i) {
            const int t   = dir ? (T_ - 1 - i) : i;
            const int off = ((b * T_ + t) << 9) + ch;
            float rv = sigm(r[off]);
            float kv = k[off];
            float vv = v[off];
            float wv = __expf(-__expf(ab + tanhf(cpr[off] * aA) * aB));
            __syncthreads();                  // protect previous iteration's reads
            kb[d] = kv; vb[d] = vv;
            __syncthreads();
            float sc = 0.0f, vs = 0.0f;
#pragma unroll
            for (int e = 0; e < D_; ++e) {
                float ke = kb[e], ve = vb[e];
                sc = fmaf(st[e], ke, sc);     // old state
                vs += ve;
                st[e] = fmaf(wv, st[e], kv * ve);
            }
            float ot = 0.5f * rv * (sc + kv * uv * vs);
            if (dir == 0) y[off] = ot; else y[off] += ot;
        }
    }
}

// ------------------------------ GroupNorm over (T,D) per (b,h) + gate + bf16 cast
__global__ void __launch_bounds__(256)
gn_gate_kernel(const float* __restrict__ y, const float* __restrict__ g,
               const float* __restrict__ gnw, const float* __restrict__ gnb,
               bf16* __restrict__ ybf) {
    const int b = blockIdx.x >> 3;
    const int h = blockIdx.x & 7;
    const int tid = threadIdx.x;
    const int NR = T_ * D_;                    // 262144
    __shared__ float s1[256], s2[256];

    float sum = 0.0f, sq = 0.0f;
    for (int i = tid; i < NR; i += 256) {
        int t = i >> 6, d = i & 63;
        float val = y[((b * T_ + t) << 9) + h * D_ + d];
        sum += val; sq += val * val;
    }
    s1[tid] = sum; s2[tid] = sq; __syncthreads();
    for (int sft = 128; sft > 0; sft >>= 1) {
        if (tid < sft) { s1[tid] += s1[tid + sft]; s2[tid] += s2[tid + sft]; }
        __syncthreads();
    }
    const float mean = s1[0] / (float)NR;
    const float var  = s2[0] / (float)NR - mean * mean;
    const float rs   = rsqrtf(var + 1e-5f);

    for (int i = tid; i < NR; i += 256) {
        int t = i >> 6, d = i & 63;
        int ch  = h * D_ + d;
        int off = ((b * T_ + t) << 9) + ch;
        float val = (y[off] - mean) * rs * gnw[ch] + gnb[ch];
        val *= sigm(g[off]);
        ybf[off] = (bf16)val;
    }
}

// --------------------------------------------------------------------- launcher
extern "C" void kernel_launch(void* const* d_in, const int* in_sizes, int n_in,
                              void* d_out, int out_size, void* d_ws, size_t ws_size,
                              hipStream_t stream) {
    (void)in_sizes; (void)n_in; (void)out_size; (void)ws_size;
    const float* x     = (const float*)d_in[0];
    const float* mur   = (const float*)d_in[1];
    const float* muk   = (const float*)d_in[2];
    const float* muv   = (const float*)d_in[3];
    const float* muw   = (const float*)d_in[4];
    const float* Wr    = (const float*)d_in[5];
    const float* Wk    = (const float*)d_in[6];
    const float* Wv    = (const float*)d_in[7];
    const float* Wg    = (const float*)d_in[8];
    const float* Ww    = (const float*)d_in[9];
    const float* dA    = (const float*)d_in[10];
    const float* dB    = (const float*)d_in[11];
    const float* dbase = (const float*)d_in[12];
    const float* u     = (const float*)d_in[13];
    const float* gnw   = (const float*)d_in[14];
    const float* gnb   = (const float*)d_in[15];
    const float* Wo    = (const float*)d_in[16];

    const size_t nW        = (size_t)C_ * C_;          // 262144
    const size_t nAct      = (size_t)BT_ * C_;         // 16777216
    const size_t WBF_BYTES = 6 * nW * sizeof(bf16);    // 3 MB
    const size_t XBF_BYTES = 5 * nAct * sizeof(bf16);  // 168 MB

    char* ws = (char*)d_ws;
    bf16*  wbf    = (bf16*)ws;                                   // 6 weights bf16
    bf16*  xstage = (bf16*)(ws + WBF_BYTES);                     // 5 activations bf16
    float* proj   = (float*)(ws + WBF_BYTES + XBF_BYTES);        // 5 projections f32

    bf16 *xr = xstage, *xk = xstage + nAct, *xv = xstage + 2 * nAct,
         *xw = xstage + 3 * nAct, *xg = xstage + 4 * nAct;
    float *pr = proj, *pk = proj + nAct, *pv = proj + 2 * nAct,
          *pc = proj + 3 * nAct, *pg = proj + 4 * nAct;
    // y / ybf reuse the xstage region (consumed by the time the scan runs)
    float* y   = (float*)xstage;
    bf16*  ybf = (bf16*)(ws + WBF_BYTES + nAct * sizeof(float));

    // 1) weights -> bf16
    dim3 gW((nW + 255) / 256);
    cvt_bf16_kernel<<<gW, 256, 0, stream>>>(Wr, wbf + 0 * nW, (int)nW);
    cvt_bf16_kernel<<<gW, 256, 0, stream>>>(Wk, wbf + 1 * nW, (int)nW);
    cvt_bf16_kernel<<<gW, 256, 0, stream>>>(Wv, wbf + 2 * nW, (int)nW);
    cvt_bf16_kernel<<<gW, 256, 0, stream>>>(Wg, wbf + 3 * nW, (int)nW);
    cvt_bf16_kernel<<<gW, 256, 0, stream>>>(Ww, wbf + 4 * nW, (int)nW);
    cvt_bf16_kernel<<<gW, 256, 0, stream>>>(Wo, wbf + 5 * nW, (int)nW);

    // 2) bqe mixes + bf16 cast
    bqe_cast_kernel<<<(unsigned)(nAct / 256), 256, 0, stream>>>(
        x, mur, muk, muv, muw, xr, xk, xv, xw, xg);

    // 3) five projection GEMMs (WMMA bf16)
    dim3 gG(BT_ / 128, C_ / 128);
    gemm_bf16_kernel<<<gG, 256, 0, stream>>>(xr, wbf + 0 * nW, pr);
    gemm_bf16_kernel<<<gG, 256, 0, stream>>>(xk, wbf + 1 * nW, pk);
    gemm_bf16_kernel<<<gG, 256, 0, stream>>>(xv, wbf + 2 * nW, pv);
    gemm_bf16_kernel<<<gG, 256, 0, stream>>>(xg, wbf + 3 * nW, pg);
    gemm_bf16_kernel<<<gG, 256, 0, stream>>>(xw, wbf + 4 * nW, pc);

    // 4) bidirectional WKV scan (fwd stores, bwd accumulates)
    wkv_scan_kernel<<<B_ * H_, 64, 0, stream>>>(pr, pk, pv, pc, dA, dB, dbase, u, y);

    // 5) group norm + sigmoid gate + bf16 cast
    gn_gate_kernel<<<B_ * H_, 256, 0, stream>>>(y, pg, gnw, gnb, ybf);

    // 6) output GEMM -> d_out (f32)
    gemm_bf16_kernel<<<gG, 256, 0, stream>>>(ybf, wbf + 5 * nW, (float*)d_out);
}
